// WindowAttention_29300266893712
// MI455X (gfx1250) — matrix operand order repaired and checked by reading, not compiled, not run on hardware
//
#include <hip/hip_runtime.h>
#include <hip/hip_bf16.h>

typedef __attribute__((ext_vector_type(16))) _Float16 v16h;
typedef __attribute__((ext_vector_type(8)))  _Float16 v8h;
typedef __attribute__((ext_vector_type(4)))  _Float16 v4h;
typedef __attribute__((ext_vector_type(8)))  float    v8f;
typedef __attribute__((ext_vector_type(4)))  float    v4f;

#define WSZ    7
#define NTOK   49
#define NP     64          // padded token count (4x 16-tiles)
#define HEADS  6
#define HD     32
#define CD     192
#define C3     576
#define QKVW_ELEMS (C3 * CD)     // 110592
#define PROJW_ELEMS (CD * CD)    // 36864

// ---- LDS layout (byte offsets into one shared buffer; x overlays S) ----
// S   : [6][64][64] f32   @ 0       (98304 B)   -- x_h [64][192] f16 overlays @0 (24576 B)
// qs  : [6][64][32] f16   @ 98304   (24576 B)
// ks  : [6][64][32] f16   @ 122880  (24576 B)
// vT  : [6][32][64] f16   @ 147456  (24576 B)
// P   : [6][64][64] f16   @ 172032  (49152 B)
// Os  : [64][192]   f16   @ 221184  (24576 B)
#define SMEM_BYTES 245760

__device__ __forceinline__ v16h frag2x8(const _Float16* p0, const _Float16* p1) {
    v8h lo = *(const v8h*)p0;
    v8h hi = *(const v8h*)p1;
    v16h r;
#pragma unroll
    for (int i = 0; i < 8; ++i) { r[i] = lo[i]; r[i + 8] = hi[i]; }
    return r;
}

__device__ __forceinline__ v8f wmma_f16(v16h a, v16h b, v8f c) {
    return __builtin_amdgcn_wmma_f32_16x16x32_f16(false, a, false, b, (short)0, c, false, false);
}

__device__ __forceinline__ int rfl(int v) { return __builtin_amdgcn_readfirstlane(v); }

// ---------------- phase 0: convert weights to f16 in workspace ----------------
__global__ void __launch_bounds__(256) cvt_weights_kernel(const float* __restrict__ qkv_w,
                                                          const float* __restrict__ proj_w,
                                                          _Float16* __restrict__ wsh) {
    int i = blockIdx.x * 256 + threadIdx.x;
    if (i < QKVW_ELEMS)                      wsh[i] = (_Float16)qkv_w[i];
    else if (i < QKVW_ELEMS + PROJW_ELEMS)   wsh[i] = (_Float16)proj_w[i - QKVW_ELEMS];
}

// ---------------- fused window attention: one block per window ----------------
__global__ void __launch_bounds__(256, 1)
win_attn_kernel(const float* __restrict__ x,
                const float* __restrict__ qkv_b,
                const float* __restrict__ proj_b,
                const float* __restrict__ bias_table,
                const _Float16* __restrict__ qkv_wh,   // [576][192] f16 row-major
                const _Float16* __restrict__ proj_wh,  // [192][192] f16 row-major
                float* __restrict__ out) {
    __shared__ __align__(32) unsigned char smem[SMEM_BYTES];
    _Float16* xs  = (_Float16*)(smem + 0);
    float*    S   = (float*)   (smem + 0);
    _Float16* qs  = (_Float16*)(smem + 98304);
    _Float16* ksm = (_Float16*)(smem + 122880);
    _Float16* vT  = (_Float16*)(smem + 147456);
    _Float16* P   = (_Float16*)(smem + 172032);
    _Float16* Os  = (_Float16*)(smem + 221184);

    const int blk = blockIdx.x;
    const int t   = threadIdx.x;
    const int w   = t >> 5;       // wave id 0..7
    const int l   = t & 31;       // lane
    const int lm  = l & 15;       // column (B/C/D) or row (A) within tile
    const int lg  = l >> 4;       // lane half-group

    // ---- Stage A: x -> LDS f16 (float4 -> 4xf16), zero-pad rows 49..63 ----
    for (int q4 = t; q4 < (NP * CD) / 4; q4 += 256) {
        const int idx = q4 * 4;
        const int m = idx / CD, c = idx % CD;
        v4h hv;
        if (m < NTOK) {
            v4f fv = *(const v4f*)(x + ((size_t)blk * NTOK + m) * CD + c);
#pragma unroll
            for (int i = 0; i < 4; ++i) hv[i] = (_Float16)fv[i];
        } else {
            hv = (v4h)(_Float16)0.0f;
        }
        *(v4h*)(xs + idx) = hv;
    }
    __syncthreads();

    // ---- Stage B: qkv = x @ qkv_w^T + qkv_b ; scatter into q/k/vT, scale q ----
    const float scale = 0.17677669529663687f;   // 1/sqrt(32)
    for (int job = w; job < 4 * 36; job += 8) {
        const int sj = rfl(job);       // wave-uniform -> SGPR
        const int ti = sj & 3;         // row tile 0..3
        const int tj = sj >> 2;        // col tile 0..35
        const int ncol  = tj * 16;
        const int which = ncol / CD;   // 0=q 1=k 2=v (scalar)
        const int h     = (ncol % CD) / HD;
        const int dbase = ncol % HD;   // 0 or 16

        const float bias = qkv_b[ncol + lm];
        v8f acc;
#pragma unroll
        for (int r = 0; r < 8; ++r) acc[r] = bias;   // fold bias into C operand

        const int rowA = ti * 16 + lm;
#pragma unroll
        for (int kc = 0; kc < 6; ++kc) {
            const _Float16* a0 = xs + rowA * CD + kc * 32 + lg * 8;
            v16h A = frag2x8(a0, a0 + 16);
            const _Float16* b0 = qkv_wh + (size_t)(ncol + lm) * CD + kc * 32 + lg * 8;
            v16h B = frag2x8(b0, b0 + 16);
            acc = wmma_f16(A, B, acc);
        }

        if (which == 0) {            // q: scale, scatter [h][m][d]
#pragma unroll
            for (int r = 0; r < 8; ++r) {
                const int m = ti * 16 + r + lg * 8;
                qs[(h * NP + m) * HD + dbase + lm] = (_Float16)(acc[r] * scale);
            }
        } else if (which == 1) {     // k: scatter [h][m][d]
#pragma unroll
            for (int r = 0; r < 8; ++r) {
                const int m = ti * 16 + r + lg * 8;
                ksm[(h * NP + m) * HD + dbase + lm] = (_Float16)acc[r];
            }
        } else {                     // v: transposed [h][d][m] -> one 16B store
            v8h hv;
#pragma unroll
            for (int r = 0; r < 8; ++r) hv[r] = (_Float16)acc[r];
            *(v8h*)(vT + (h * HD + dbase + lm) * NP + ti * 16 + lg * 8) = hv;
        }
    }
    __syncthreads();

    // ---- Stage C: S = q @ k^T  (hd=32 -> one WMMA per 16x16 tile) ----
    for (int job = w; job < HEADS * 16; job += 8) {
        const int sj = rfl(job);
        const int h  = sj >> 4;
        const int ti = (sj >> 2) & 3;
        const int tj = sj & 3;
        const _Float16* a0 = qs + (h * NP + ti * 16 + lm) * HD + lg * 8;
        v16h A = frag2x8(a0, a0 + 16);
        const _Float16* b0 = ksm + (h * NP + tj * 16 + lm) * HD + lg * 8;
        v16h B = frag2x8(b0, b0 + 16);
        v8f acc = {0.f, 0.f, 0.f, 0.f, 0.f, 0.f, 0.f, 0.f};
        acc = wmma_f16(A, B, acc);
#pragma unroll
        for (int r = 0; r < 8; ++r) {
            const int m = ti * 16 + r + lg * 8;
            S[(h * NP + m) * NP + tj * 16 + lm] = acc[r];
        }
    }
    __syncthreads();

    // ---- Stage D: softmax rows with relative-position bias; P f16, padded cols = 0 ----
    for (int jr = t; jr < HEADS * NP; jr += 256) {
        const int h = jr / NP, i = jr % NP;
        _Float16* prow = P + (size_t)(h * NP + i) * NP;
        if (i >= NTOK) {
#pragma unroll
            for (int j4 = 0; j4 < NP / 4; ++j4) *(v4h*)(prow + j4 * 4) = (v4h)(_Float16)0.0f;
            continue;
        }
        float* srow = S + (size_t)(h * NP + i) * NP;
        const int iy = i / WSZ, ix = i % WSZ;
        float mx = -1e30f;
#pragma unroll
        for (int j = 0; j < NTOK; ++j) {
            const int jy = j / WSZ, jx = j % WSZ;
            const int ridx = (iy - jy + WSZ - 1) * (2 * WSZ - 1) + (ix - jx + WSZ - 1);
            const float s = srow[j] + bias_table[ridx * HEADS + h];
            srow[j] = s;
            mx = fmaxf(mx, s);
        }
        float sum = 0.0f;
#pragma unroll
        for (int j = 0; j < NTOK; ++j) {
            const float e = __expf(srow[j] - mx);
            srow[j] = e;
            sum += e;
        }
        const float inv = 1.0f / sum;
#pragma unroll
        for (int j = 0; j < NTOK; ++j) prow[j] = (_Float16)(srow[j] * inv);
#pragma unroll
        for (int j = NTOK; j < NP; ++j) prow[j] = (_Float16)0.0f;
    }
    __syncthreads();

    // ---- Stage E: O = P @ V (per head), store O as [64][192] f16 ----
    for (int job = w; job < HEADS * 8; job += 8) {
        const int sj = rfl(job);
        const int h  = sj >> 3;
        const int ti = (sj >> 1) & 3;
        const int tj = sj & 1;
        v8f acc = {0.f, 0.f, 0.f, 0.f, 0.f, 0.f, 0.f, 0.f};
#pragma unroll
        for (int kc = 0; kc < 2; ++kc) {
            const _Float16* a0 = P + (size_t)(h * NP + ti * 16 + lm) * NP + kc * 32 + lg * 8;
            v16h A = frag2x8(a0, a0 + 16);
            const _Float16* b0 = vT + (size_t)(h * HD + tj * 16 + lm) * NP + kc * 32 + lg * 8;
            v16h B = frag2x8(b0, b0 + 16);
            acc = wmma_f16(A, B, acc);
        }
#pragma unroll
        for (int r = 0; r < 8; ++r) {
            const int m = ti * 16 + r + lg * 8;
            Os[m * CD + h * HD + tj * 16 + lm] = (_Float16)acc[r];
        }
    }
    __syncthreads();

    // ---- Stage F: out = O @ proj_w^T + proj_b (f32 store, rows < 49) ----
    for (int job = w; job < 4 * 12; job += 8) {
        const int sj = rfl(job);
        const int ti = sj / 12;
        const int tj = sj % 12;
        const float bias = proj_b[tj * 16 + lm];
        v8f acc;
#pragma unroll
        for (int r = 0; r < 8; ++r) acc[r] = bias;   // fold bias into C operand
#pragma unroll
        for (int kc = 0; kc < 6; ++kc) {
            const _Float16* a0 = Os + (ti * 16 + lm) * CD + kc * 32 + lg * 8;
            v16h A = frag2x8(a0, a0 + 16);
            const _Float16* b0 = proj_wh + (size_t)(tj * 16 + lm) * CD + kc * 32 + lg * 8;
            v16h B = frag2x8(b0, b0 + 16);
            acc = wmma_f16(A, B, acc);
        }
#pragma unroll
        for (int r = 0; r < 8; ++r) {
            const int m = ti * 16 + r + lg * 8;
            if (m < NTOK)
                out[((size_t)blk * NTOK + m) * CD + tj * 16 + lm] = acc[r];
        }
    }
}

extern "C" void kernel_launch(void* const* d_in, const int* in_sizes, int n_in,
                              void* d_out, int out_size, void* d_ws, size_t ws_size,
                              hipStream_t stream) {
    const float* x          = (const float*)d_in[0];
    const float* qkv_w      = (const float*)d_in[1];
    const float* qkv_b      = (const float*)d_in[2];
    const float* proj_w     = (const float*)d_in[3];
    const float* proj_b     = (const float*)d_in[4];
    const float* bias_table = (const float*)d_in[5];
    float* out = (float*)d_out;

    _Float16* wsh = (_Float16*)d_ws;           // [qkv_w f16 | proj_w f16]
    const int B = in_sizes[0] / (NTOK * CD);   // 8192 windows

    const int cvt_total = QKVW_ELEMS + PROJW_ELEMS;
    cvt_weights_kernel<<<(cvt_total + 255) / 256, 256, 0, stream>>>(qkv_w, proj_w, wsh);

    win_attn_kernel<<<B, 256, 0, stream>>>(x, qkv_b, proj_b, bias_table,
                                           wsh, wsh + QKVW_ELEMS, out);
}
